// ContextSim_90812788506738
// MI455X (gfx1250) — matrix operand order repaired and checked by reading, not compiled, not run on hardware
//
#include <hip/hip_runtime.h>
#include <stdint.h>

// ---------------- CDNA5 WMMA types / fragment helpers ----------------
typedef __attribute__((ext_vector_type(16))) __bf16 v16bf;
typedef __attribute__((ext_vector_type(8)))  float  v8f;

union FragBF { v16bf v; uint32_t u[8]; };

__device__ __forceinline__ uint16_t f2bf(float f) {
  uint32_t u = __float_as_uint(f);
  u += 0x7FFFu + ((u >> 16) & 1u);     // round-to-nearest-even
  return (uint16_t)(u >> 16);
}

// A fragment: 16 x K row-major bf16 (rows m0..m0+15 at base), K-slice [k0,k0+32)
// Layout per ISA 7.12.2: lanes 0-15 row M, K={0..7,16..23}; lanes 16-31 K+=8.
__device__ __forceinline__ v16bf load_frag_a(const uint16_t* __restrict__ A,
                                             int lda, int k0, int lane) {
  const int half = lane >> 4, m = lane & 15;
  const uint16_t* r = A + (size_t)m * lda + k0 + (half ? 8 : 0);
  FragBF f;
#pragma unroll
  for (int j = 0; j < 4; ++j) {
    f.u[j]     = *(const uint32_t*)(r + 2 * j);
    f.u[j + 4] = *(const uint32_t*)(r + 16 + 2 * j);
  }
  return f.v;
}

// B fragment: logical B[k][n] = W[n][k], W row-major [N,K] bf16; base = W + n0*ldw.
// Lanes 0-15 hold column n with K={0..7,16..23}, lanes 16-31 K+=8 (2 per dword).
__device__ __forceinline__ v16bf load_frag_b(const uint16_t* __restrict__ Wn0,
                                             int ldw, int k0, int lane) {
  const int half = lane >> 4, n = lane & 15;
  const uint16_t* r = Wn0 + (size_t)n * ldw + k0 + 8 * half;
  FragBF f;
#pragma unroll
  for (int j = 0; j < 4; ++j) {
    f.u[j]     = *(const uint32_t*)(r + 2 * j);
    f.u[j + 4] = *(const uint32_t*)(r + 16 + 2 * j);
  }
  return f.v;
}

// ---------------- generic bf16 WMMA GEMM: C = A @ W^T + bias ----------------
// A: [M,K] bf16 row-major; W: [N,K] bf16 row-major; C: f32, ldc.
// Register-blocked: each wave computes a 64x16 strip (4 M-tiles x 1 N-tile),
// reusing each B fragment across 4 WMMAs (4x less weight traffic from L2).
// M must be a multiple of 64 (here M == 1024 always).
// row_map==1 remaps output row m = s*8+b -> b*128+s (seq-major -> batch-major).
__global__ void wmma_gemm_bias(const uint16_t* __restrict__ A, int lda,
                               const uint16_t* __restrict__ W, int ldw,
                               const float* __restrict__ bias,
                               float* __restrict__ C, int ldc,
                               int N, int K, int row_map) {
  const int lane  = threadIdx.x & 31;
  const int wave  = threadIdx.x >> 5;
  const int ntile = blockIdx.x * 8 + wave;
  if (ntile * 16 >= N) return;
  const int n0 = ntile * 16;
  const int m0 = blockIdx.y * 64;
  const uint16_t* Ab = A + (size_t)m0 * lda;
  const uint16_t* Wb = W + (size_t)n0 * ldw;
  v8f acc0 = {}, acc1 = {}, acc2 = {}, acc3 = {};
  const uint16_t* pfb = Wb + (size_t)(lane & 15) * ldw;
  for (int k0 = 0; k0 < K; k0 += 32) {
    // prefetch the streaming weight rows a few K-slices ahead (global_prefetch_b8)
    __builtin_prefetch(pfb + k0 + 128, 0, 1);
    v16bf b  = load_frag_b(Wb, ldw, k0, lane);
    v16bf a0 = load_frag_a(Ab,                        lda, k0, lane);
    v16bf a1 = load_frag_a(Ab + (size_t)16 * lda,     lda, k0, lane);
    v16bf a2 = load_frag_a(Ab + (size_t)32 * lda,     lda, k0, lane);
    v16bf a3 = load_frag_a(Ab + (size_t)48 * lda,     lda, k0, lane);
    acc0 = __builtin_amdgcn_wmma_f32_16x16x32_bf16(false, a0, false, b, (short)0, acc0, false, false);
    acc1 = __builtin_amdgcn_wmma_f32_16x16x32_bf16(false, a1, false, b, (short)0, acc1, false, false);
    acc2 = __builtin_amdgcn_wmma_f32_16x16x32_bf16(false, a2, false, b, (short)0, acc2, false, false);
    acc3 = __builtin_amdgcn_wmma_f32_16x16x32_bf16(false, a3, false, b, (short)0, acc3, false, false);
  }
  const int n = lane & 15, half = lane >> 4;
  const float bv = bias ? bias[n0 + n] : 0.0f;
  v8f accs[4] = {acc0, acc1, acc2, acc3};
#pragma unroll
  for (int i = 0; i < 4; ++i) {
#pragma unroll
    for (int v = 0; v < 8; ++v) {
      int m   = m0 + i * 16 + v + 8 * half;
      int row = row_map ? ((m & 7) * 128 + (m >> 3)) : m;
      C[(size_t)row * ldc + n0 + n] = accs[i][v] + bv;
    }
  }
}

// ---------------- fused GRU recurrent step (one timestep, one layer) --------
// h_prev: bf16 [16, dh] (rows 8..15 zero) + f32 [8, dh].  Computes
// gh = h_prev @ Whh^T for gate columns [u0,u0+16) of all three gates via WMMA
// (one A fragment feeds 3 WMMAs), then gate math, writing h_new (ping-pong)
// and the layer output (bf16).
__global__ void gru_step(const uint16_t* __restrict__ hprev_bf,
                         const float*    __restrict__ hprev_f,
                         uint16_t*       __restrict__ hnew_bf,
                         float*          __restrict__ hnew_f,
                         const float*    __restrict__ gi,    // [1024, 3*dh]
                         const uint16_t* __restrict__ whh,   // [3*dh, dh] bf16
                         const float*    __restrict__ bhh,   // [3*dh]
                         uint16_t*       __restrict__ xout,  // [1024, dh] bf16
                         int dh, int t) {
  const int lane = threadIdx.x & 31;
  const int u0   = blockIdx.x * 16;
  v8f ar = {}, az = {}, an = {};
  for (int k0 = 0; k0 < dh; k0 += 32) {
    v16bf a  = load_frag_a(hprev_bf, dh, k0, lane);
    v16bf br = load_frag_b(whh + (size_t)(0 * dh + u0) * dh, dh, k0, lane);
    v16bf bz = load_frag_b(whh + (size_t)(1 * dh + u0) * dh, dh, k0, lane);
    v16bf bn = load_frag_b(whh + (size_t)(2 * dh + u0) * dh, dh, k0, lane);
    ar = __builtin_amdgcn_wmma_f32_16x16x32_bf16(false, a, false, br, (short)0, ar, false, false);
    az = __builtin_amdgcn_wmma_f32_16x16x32_bf16(false, a, false, bz, (short)0, az, false, false);
    an = __builtin_amdgcn_wmma_f32_16x16x32_bf16(false, a, false, bn, (short)0, an, false, false);
  }
  const int n = lane & 15, half = lane >> 4;
  if (half == 0) {                      // lanes 0..15 hold batch rows 0..7
    const int u = u0 + n;
    const float bhr = bhh[u], bhz = bhh[dh + u], bhn = bhh[2 * dh + u];
#pragma unroll
    for (int v = 0; v < 8; ++v) {
      const size_t row = (size_t)t * 8 + v;
      const float* gir = gi + row * (size_t)(3 * dh);
      float r  = 1.0f / (1.0f + __expf(-(gir[u]          + ar[v] + bhr)));
      float z  = 1.0f / (1.0f + __expf(-(gir[dh + u]     + az[v] + bhz)));
      float nn = tanhf(gir[2 * dh + u] + r * (an[v] + bhn));
      float hp = hprev_f[(size_t)v * dh + u];
      float h  = (1.0f - z) * nn + z * hp;
      hnew_f[(size_t)v * dh + u]  = h;
      uint16_t hb = f2bf(h);
      hnew_bf[(size_t)v * dh + u] = hb;
      xout[row * dh + u] = hb;
    }
  }
}

// ---------------- embedding gather: ctx (B-major f32) + seq-major bf16 ------
__global__ void gather_embed(const float* __restrict__ E, const int* __restrict__ gidx,
                             uint16_t* __restrict__ emb_bf, float* __restrict__ ctx) {
  const int blk = blockIdx.x;            // = b*128 + s
  const int b = blk >> 7, s = blk & 127;
  const int idx = gidx[blk];
  const float* src = E + (size_t)idx * 512;
  float* cdst = ctx + (size_t)blk * 512;
  uint16_t* edst = emb_bf + (size_t)(s * 8 + b) * 512;
  for (int d = threadIdx.x; d < 512; d += blockDim.x) {
    float v = src[d];
    cdst[d] = v * 0.05f;                 // /C, C=20
    edst[d] = f2bf(v);
  }
}

__global__ void conv_bf16(const float* __restrict__ s, uint16_t* __restrict__ d, size_t n) {
  size_t i  = (size_t)blockIdx.x * blockDim.x + threadIdx.x;
  size_t st = (size_t)gridDim.x * blockDim.x;
  for (; i < n; i += st) d[i] = f2bf(s[i]);
}

__global__ void fill_f32(float* __restrict__ p, float v, size_t n) {
  size_t i  = (size_t)blockIdx.x * blockDim.x + threadIdx.x;
  size_t st = (size_t)gridDim.x * blockDim.x;
  for (; i < n; i += st) p[i] = v;
}

// ---------------- in-place log_softmax + top-5 per row ----------------------
__global__ void logsoftmax_topk(float* __restrict__ logits, int* __restrict__ kidx, int N) {
  __shared__ float sred[256];
  __shared__ float stv[256 * 5];
  __shared__ int   sti[256 * 5];
  __shared__ float smax, slog;
  const int row = blockIdx.x;
  float* rp = logits + (size_t)row * N;
  float tv[5]; int ti[5];
#pragma unroll
  for (int i = 0; i < 5; ++i) { tv[i] = -INFINITY; ti[i] = 0x7fffffff; }
  for (int j = threadIdx.x; j < N; j += 256) {
    float x = rp[j];
    if (x > tv[4] || (x == tv[4] && j < ti[4])) {
      int p = 4;
      while (p > 0 && (x > tv[p - 1] || (x == tv[p - 1] && j < ti[p - 1]))) {
        tv[p] = tv[p - 1]; ti[p] = ti[p - 1]; --p;
      }
      tv[p] = x; ti[p] = j;
    }
  }
#pragma unroll
  for (int i = 0; i < 5; ++i) { stv[threadIdx.x * 5 + i] = tv[i]; sti[threadIdx.x * 5 + i] = ti[i]; }
  __syncthreads();
  if (threadIdx.x == 0) {
    for (int k = 0; k < 5; ++k) {
      float best = -INFINITY; int bi = 0x7fffffff, bp = -1;
      for (int p = 0; p < 256 * 5; ++p)
        if (stv[p] > best || (stv[p] == best && sti[p] < bi)) { best = stv[p]; bi = sti[p]; bp = p; }
      kidx[row * 5 + k] = (bi == 0x7fffffff) ? 0 : bi;
      if (bp >= 0) stv[bp] = -INFINITY;
      if (k == 0) smax = best;
    }
  }
  __syncthreads();
  const float mx = smax;
  float lsum = 0.0f;
  for (int j = threadIdx.x; j < N; j += 256) lsum += __expf(rp[j] - mx);
  sred[threadIdx.x] = lsum; __syncthreads();
  for (int off = 128; off > 0; off >>= 1) {
    if (threadIdx.x < off) sred[threadIdx.x] += sred[threadIdx.x + off];
    __syncthreads();
  }
  if (threadIdx.x == 0) slog = logf(sred[0]);
  __syncthreads();
  const float ls = slog;
  for (int j = threadIdx.x; j < N; j += 256) rp[j] = rp[j] - mx - ls;
}

// ---------------- sense branch: cosine argmax over K*32 candidates ----------
__global__ void senses_kernel(const float* __restrict__ ctx, const float* __restrict__ X,
                              const int* __restrict__ kidx, const int* __restrict__ grapharea,
                              float* __restrict__ pred_senses) {
  __shared__ __align__(16) float c[512];
  __shared__ float cred[256];
  __shared__ float scos[160];
  __shared__ int   sidx[160];
  const int row = blockIdx.x;
  const float* cp = ctx + (size_t)row * 512;
  float nrm = 0.0f;
  for (int d = threadIdx.x; d < 512; d += 256) { float v = cp[d]; c[d] = v; nrm += v * v; }
  cred[threadIdx.x] = nrm; __syncthreads();
  for (int off = 128; off > 0; off >>= 1) {
    if (threadIdx.x < off) cred[threadIdx.x] += cred[threadIdx.x + off];
    __syncthreads();
  }
  const float cnorm = sqrtf(cred[0]);
  const int j = threadIdx.x;
  if (j < 160) {
    const int k = j >> 5, g = j & 31;
    const int gr = kidx[row * 5 + k] + 70000;                // + NSENSES
    const int nb = grapharea[(size_t)gr * 32 + g] - 1;
    const bool valid = (nb > 0) && (nb < 70000);
    const int safe = valid ? nb : 0;
    const float* xr = X + (size_t)safe * 512;
    float dot = 0.0f, n2 = 0.0f;
    for (int d = 0; d < 512; d += 4) {
      float4 xv = *(const float4*)(xr + d);
      float4 cv = *(const float4*)(c + d);
      dot += xv.x * cv.x + xv.y * cv.y + xv.z * cv.z + xv.w * cv.w;
      n2  += xv.x * xv.x + xv.y * xv.y + xv.z * xv.z + xv.w * xv.w;
    }
    const float den = fmaxf(cnorm * sqrtf(n2), 1e-8f);
    scos[j] = valid ? dot / den : -INFINITY;
    sidx[j] = safe;
  }
  __syncthreads();
  if (threadIdx.x == 0) {
    float best = -INFINITY; int bi = sidx[0];
    for (int p = 0; p < 160; ++p)
      if (scos[p] > best) { best = scos[p]; bi = sidx[p]; }
    pred_senses[(size_t)row * 70000 + bi] = logf(1.0f - 1e-8f * 69999.0f);
  }
}

// ---------------- host-side orchestration -----------------------------------
extern "C" void kernel_launch(void* const* d_in, const int* in_sizes, int n_in,
                              void* d_out, int out_size, void* d_ws, size_t ws_size,
                              hipStream_t stream) {
  (void)in_sizes; (void)n_in; (void)out_size; (void)ws_size;
  const float* E  = (const float*)d_in[0];
  const float* X  = (const float*)d_in[1];
  const float* Wih[3] = {(const float*)d_in[2], (const float*)d_in[6],  (const float*)d_in[10]};
  const float* Whh[3] = {(const float*)d_in[3], (const float*)d_in[7],  (const float*)d_in[11]};
  const float* bih[3] = {(const float*)d_in[4], (const float*)d_in[8],  (const float*)d_in[12]};
  const float* bhh[3] = {(const float*)d_in[5], (const float*)d_in[9],  (const float*)d_in[13]};
  const float* Wg = (const float*)d_in[14];
  const float* bg = (const float*)d_in[15];
  const int* gidx = (const int*)d_in[16];
  const int* grapharea = (const int*)d_in[17];

  float* pred_globals = (float*)d_out;                          // [1024, 40000]
  float* pred_senses  = (float*)d_out + (size_t)1024 * 40000;   // [1024, 70000]

  char* wp = (char*)d_ws;
  auto carve = [&](size_t bytes) -> char* {
    char* p = wp; wp += (bytes + 255) & ~(size_t)255; return p;
  };
  uint16_t* emb_bf = (uint16_t*)carve((size_t)1024 * 512 * 2);
  float*    ctx    = (float*)   carve((size_t)1024 * 512 * 4);
  const size_t wihN[3] = {3072 * 512, 3072 * 1024, 1536 * 1024};
  const size_t whhN[3] = {3072 * 1024, 3072 * 1024, 1536 * 512};
  uint16_t* wihb[3]; uint16_t* whhb[3];
  for (int l = 0; l < 3; ++l) {
    wihb[l] = (uint16_t*)carve(wihN[l] * 2);
    whhb[l] = (uint16_t*)carve(whhN[l] * 2);
  }
  uint16_t* wgb   = (uint16_t*)carve((size_t)40000 * 512 * 2);
  float*    gi    = (float*)   carve((size_t)1024 * 3072 * 4);
  uint16_t* xout0 = (uint16_t*)carve((size_t)1024 * 1024 * 2);
  uint16_t* xout1 = (uint16_t*)carve((size_t)1024 * 1024 * 2);
  uint16_t* xout2 = (uint16_t*)carve((size_t)1024 * 512 * 2);
  uint16_t* hb_bf = (uint16_t*)carve((size_t)2 * 16 * 1024 * 2);  // ping-pong
  float*    hb_f  = (float*)   carve((size_t)2 * 8 * 1024 * 4);
  int*      kidx  = (int*)     carve((size_t)1024 * 5 * 4);

  // 1. weights -> bf16
  for (int l = 0; l < 3; ++l) {
    conv_bf16<<<1024, 256, 0, stream>>>(Wih[l], wihb[l], wihN[l]);
    conv_bf16<<<1024, 256, 0, stream>>>(Whh[l], whhb[l], whhN[l]);
  }
  conv_bf16<<<4096, 256, 0, stream>>>(Wg, wgb, (size_t)40000 * 512);

  // 2. embedding gather (ctx B-major f32; GRU input seq-major bf16)
  gather_embed<<<1024, 128, 0, stream>>>(E, gidx, emb_bf, ctx);

  // 3. GRU stack
  const int dinL[3] = {512, 1024, 1024};
  const int dhL[3]  = {1024, 1024, 512};
  uint16_t* xouts[3] = {xout0, xout1, xout2};
  const uint16_t* xin = emb_bf;
  for (int l = 0; l < 3; ++l) {
    const int dh = dhL[l], din = dinL[l], N3 = 3 * dh;
    fill_f32<<<64, 256, 0, stream>>>((float*)hb_bf, 0.0f, (size_t)2 * 16 * 1024 / 2);
    fill_f32<<<64, 256, 0, stream>>>(hb_f, 0.0f, (size_t)2 * 8 * 1024);
    // batched input transform: gi = X_in @ Wih^T + bih   (WMMA, M=1024)
    wmma_gemm_bias<<<dim3((N3 / 16 + 7) / 8, 16), 256, 0, stream>>>(
        xin, din, wihb[l], din, bih[l], gi, N3, N3, din, 0);
    // sequential recurrence
    for (int t = 0; t < 128; ++t) {
      const int rb = t & 1, wb = rb ^ 1;
      gru_step<<<dh / 16, 32, 0, stream>>>(
          hb_bf + (size_t)rb * 16 * 1024, hb_f + (size_t)rb * 8 * 1024,
          hb_bf + (size_t)wb * 16 * 1024, hb_f + (size_t)wb * 8 * 1024,
          gi, whhb[l], bhh[l], xouts[l], dh, t);
    }
    xin = xouts[l];
  }

  // 4. logits = h @ Wg^T + bg, output batch-major straight into d_out
  wmma_gemm_bias<<<dim3((40000 / 16 + 7) / 8, 16), 256, 0, stream>>>(
      xout2, 512, wgb, 512, bg, pred_globals, 40000, 40000, 512, 1);

  // 5. in-place log_softmax + top-5
  logsoftmax_topk<<<1024, 256, 0, stream>>>(pred_globals, kidx, 40000);

  // 6. pred_senses: fill log(EPS), then scatter selected sense per row
  fill_f32<<<4096, 256, 0, stream>>>(pred_senses, -18.420680743952367f,
                                     (size_t)1024 * 70000);
  senses_kernel<<<1024, 256, 0, stream>>>(ctx, X, kidx, grapharea, pred_senses);
}
